// DeepTrader_13408887898598
// MI455X (gfx1250) — matrix operand order repaired and checked by reading, not compiled, not run on hardware
//
#include <hip/hip_runtime.h>

// ---------------------------------------------------------------------------
// DeepTrader forward for MI455X (gfx1250, wave32).
// All dense contractions run on v_wmma_f32_16x16x32_bf16 (bf16 in, f32 acc).
// Activations kept in bf16 mirrors whose innermost dim == GEMM K dimension,
// so each WMMA operand is fed by two global_load_b128 per lane.
// ---------------------------------------------------------------------------

typedef __attribute__((ext_vector_type(16))) __bf16 v16bf;
typedef __attribute__((ext_vector_type(8)))  float  v8f;

__device__ __forceinline__ unsigned short f2bf(float f) {
  unsigned u = __float_as_uint(f);
  u += 0x7FFFu + ((u >> 16) & 1u);          // round-to-nearest-even
  return (unsigned short)(u >> 16);
}

// ---------------------------------------------------------------------------
// WMMA GEMM:  D[i,j] = (acc? D:0) + bias[i] + sum_k A[i,k]*B[k,j]
//   A (bf16):  A[i*lda + k]                       (K contiguous)
//   B (bf16):  B[(j/bJD)*bJO + (j%bJD)*bJI + bJS + k]   (K contiguous)
//   D (f32) :  D[i*dRS + (j/dJD)*dJO + (j%dJD)*dJI + dJS]
// One 16x64 output strip per wave (A reused across 4 WMMAs).
// K % 32 == 0, Mrows % 16 == 0, Ncols % 64 == 0 at every call site.
// ---------------------------------------------------------------------------
__global__ __launch_bounds__(128) void k_wmma_gemm(
    float* __restrict__ D, const unsigned short* __restrict__ A,
    const unsigned short* __restrict__ Bm,
    int Mrows, int Ncols, int K, long lda,
    int bJD, long bJO, long bJI, long bJS,
    long dRS, int dJD, long dJO, long dJI, long dJS,
    const float* __restrict__ bias, int accum, int act) {
  int wave = blockIdx.x * (blockDim.x >> 5) + (threadIdx.x >> 5);
  int lane = threadIdx.x & 31;
  int tilesM = Mrows >> 4;
  int tilesN = Ncols >> 6;                       // 64 columns per wave
  if (wave >= tilesM * tilesN) return;           // wave-uniform: EXEC all-1s
  int tm = wave % tilesM;
  int tn = wave / tilesM;
  int rowBase = tm << 4;
  int colBase = tn << 6;

  int half = lane >> 4;                          // 0: lanes 0-15, 1: 16-31
  int aRow = rowBase + (lane & 15);
  const unsigned short* aP = A + (long)aRow * lda;

  long cbB[4], cbD[4];
#pragma unroll
  for (int q = 0; q < 4; ++q) {
    int j = colBase + (q << 4) + (lane & 15);
    cbB[q] = (long)(j / bJD) * bJO + (long)(j % bJD) * bJI + bJS;
    cbD[q] = (long)(j / dJD) * dJO + (long)(j % dJD) * dJI + dJS;
  }

  union U16 { uint4 q2[2]; v16bf v; };
  v8f acc[4] = {{0.f,0.f,0.f,0.f,0.f,0.f,0.f,0.f},
                {0.f,0.f,0.f,0.f,0.f,0.f,0.f,0.f},
                {0.f,0.f,0.f,0.f,0.f,0.f,0.f,0.f},
                {0.f,0.f,0.f,0.f,0.f,0.f,0.f,0.f}};
  const int klo = half << 3;                     // A K-offset per ISA layout
  const int kbo = half << 3;                     // B K-offset per ISA layout
  for (int k0 = 0; k0 < K; k0 += 32) {
    U16 a;
    a.q2[0] = *(const uint4*)(aP + k0 + klo);          // K = klo..klo+7
    a.q2[1] = *(const uint4*)(aP + k0 + klo + 16);     // K = klo+16..klo+23
#pragma unroll
    for (int q = 0; q < 4; ++q) {
      U16 b;
      const unsigned short* bP = Bm + cbB[q] + k0 + kbo;
      b.q2[0] = *(const uint4*)(bP);                   // K = kbo..kbo+7
      b.q2[1] = *(const uint4*)(bP + 16);              // K = kbo+16..kbo+23
      acc[q] = __builtin_amdgcn_wmma_f32_16x16x32_bf16(
          false, a.v, false, b.v, (short)0, acc[q], false, false);
    }
  }

#pragma unroll
  for (int q = 0; q < 4; ++q) {
    float v[8];
#pragma unroll
    for (int r = 0; r < 8; ++r) {
      int row = rowBase + r + (half << 3);
      v[r] = acc[q][r];
      if (bias) v[r] += bias[row];
    }
    if (dRS == 1) {                               // channel/node-last: b128 stores
      float* dp = D + cbD[q] + rowBase + (half << 3);
      if (accum) {
        float4 o0 = *(const float4*)(dp);
        float4 o1 = *(const float4*)(dp + 4);
        v[0]+=o0.x; v[1]+=o0.y; v[2]+=o0.z; v[3]+=o0.w;
        v[4]+=o1.x; v[5]+=o1.y; v[6]+=o1.z; v[7]+=o1.w;
      }
      if (act == 1) {
#pragma unroll
        for (int r = 0; r < 8; ++r) v[r] = fmaxf(v[r], 0.f);
      }
      *(float4*)(dp)     = make_float4(v[0], v[1], v[2], v[3]);
      *(float4*)(dp + 4) = make_float4(v[4], v[5], v[6], v[7]);
    } else {
#pragma unroll
      for (int r = 0; r < 8; ++r) {
        int row = rowBase + r + (half << 3);
        long idx = (long)row * dRS + cbD[q];
        float w = v[r];
        if (accum) w += D[idx];
        if (act == 1) w = fmaxf(w, 0.f);
        D[idx] = w;
      }
    }
  }
}

// ---------------------------------------------------------------------------
// BatchNorm (train-mode), channel-last layout [P, 128]: chunked coalesced
// partial sums + atomics, then finalize, then apply (+relu/tanh, bf16 mirrors).
// ---------------------------------------------------------------------------
#define BNCHUNK 2048

__global__ void k_zero256(float* __restrict__ p) { p[threadIdx.x] = 0.f; }

__global__ void k_bncl_stats(const float* __restrict__ x, long P, float* __restrict__ sums) {
  int c = threadIdx.x & 127;
  int sub = threadIdx.x >> 7;                    // 0..1
  long p0 = (long)blockIdx.x * BNCHUNK;
  long p1 = p0 + BNCHUNK; if (p1 > P) p1 = P;
  float s = 0.f, ss = 0.f;
  for (long p = p0 + sub; p < p1; p += 2) {
    float v = x[p * 128 + c]; s += v; ss += v * v;
  }
  __shared__ float sh[256];
  sh[threadIdx.x] = s; __syncthreads();
  if (sub == 0) atomicAdd(&sums[c], sh[c] + sh[128 + c]);
  __syncthreads();
  sh[threadIdx.x] = ss; __syncthreads();
  if (sub == 0) atomicAdd(&sums[128 + c], sh[c] + sh[128 + c]);
}

__global__ void k_bn_fin(const float* __restrict__ sums, long P, float* __restrict__ mv) {
  int c = threadIdx.x;
  float m = sums[c] / (float)P;
  float v = sums[128 + c] / (float)P - m * m;
  mv[c] = m;
  mv[128 + c] = rsqrtf(v + 1e-5f);
}

// x channel-last [P,128]; optional bf16 mirrors: channel-last and node-last
__global__ void k_bncl_apply(float* __restrict__ y, const float* __restrict__ x,
                             const float* __restrict__ mv,
                             const float* __restrict__ g, const float* __restrict__ b,
                             int L, long total, int act,
                             unsigned short* __restrict__ bfcl,
                             unsigned short* __restrict__ bfnl) {
  long i = blockIdx.x * (long)blockDim.x + threadIdx.x;
  if (i >= total) return;
  int c = (int)(i & 127);
  float v = (x[i] - mv[c]) * mv[128 + c] * g[c] + b[c];
  if (act == 1) v = fmaxf(v, 0.f);
  else if (act == 2) v = tanhf(v);
  y[i] = v;
  unsigned short h = f2bf(v);
  if (bfcl) bfcl[i] = h;
  if (bfnl) {                                    // p = (b*1024+n)*L + l
    long p = i >> 7;
    long bb = p / (1024L * L);
    long rem = p - bb * 1024L * L;
    long n = rem / L;
    long l = rem - n * L;
    bfnl[(((long)c * 8 + bb) * L + l) * 1024 + n] = h;
  }
}

// channel-major BN (contiguous per channel), used for the small attention norms
__global__ void k_bncm_stats(const float* __restrict__ x, long M,
                             float* __restrict__ mean, float* __restrict__ rstd) {
  int c = blockIdx.x;
  const float* p = x + (long)c * M;
  float s = 0.f, ss = 0.f;
  for (long i = threadIdx.x; i < M; i += blockDim.x) { float v = p[i]; s += v; ss += v * v; }
  __shared__ float s1[256], s2[256];
  s1[threadIdx.x] = s; s2[threadIdx.x] = ss; __syncthreads();
  for (int o = 128; o > 0; o >>= 1) {
    if ((int)threadIdx.x < o) { s1[threadIdx.x] += s1[threadIdx.x + o];
                                s2[threadIdx.x] += s2[threadIdx.x + o]; }
    __syncthreads();
  }
  if (threadIdx.x == 0) {
    float m = s1[0] / (float)M;
    float v = s2[0] / (float)M - m * m;
    mean[c] = m; rstd[c] = rsqrtf(v + 1e-5f);
  }
}

__global__ void k_bncm_apply(float* __restrict__ y, const float* __restrict__ x,
                             const float* __restrict__ mean, const float* __restrict__ rstd,
                             const float* __restrict__ g, const float* __restrict__ b,
                             long M, long total, unsigned short* __restrict__ bf) {
  long i = blockIdx.x * (long)blockDim.x + threadIdx.x;
  if (i >= total) return;
  long c = i / M;
  float v = (x[i] - mean[c]) * rstd[c] * g[c] + b[c];
  y[i] = v;
  if (bf) bf[i] = f2bf(v);
}

// ---------------------------------------------------------------------------
// Conversions
// ---------------------------------------------------------------------------
__global__ void k_f2bf(unsigned short* __restrict__ dst, const float* __restrict__ src,
                       long n, int stride) {
  long i = blockIdx.x * (long)blockDim.x + threadIdx.x;
  if (i < n) dst[i] = f2bf(src[i * (long)stride]);
}

// node-last f32 -> node-last bf16 (optional) + channel-last bf16
__global__ void k_cvt_nl(const float* __restrict__ src, unsigned short* __restrict__ dnl,
                         unsigned short* __restrict__ dcl, int L, long total) {
  long i = blockIdx.x * (long)blockDim.x + threadIdx.x;
  if (i >= total) return;
  unsigned short h = f2bf(src[i]);
  if (dnl) dnl[i] = h;
  long n = i & 1023;
  long q = i >> 10;                              // (c*8+b)*L + l
  long l = q % L;
  long cb = q / L;
  long b = cb & 7;
  long c = cb >> 3;
  dcl[(((b << 10) + n) * L + l) * 128 + c] = h;
}

// channel-last f32 -> node-last bf16
__global__ void k_cvt_cl2nl(const float* __restrict__ src, unsigned short* __restrict__ dst,
                            int L, long total) {
  long i = blockIdx.x * (long)blockDim.x + threadIdx.x;
  if (i >= total) return;
  long c = i & 127;
  long p = i >> 7;                               // (b*1024+n)*L + l
  long bb = p / (1024L * L);
  long rem = p - bb * 1024L * L;
  long n = rem / L;
  long l = rem - n * L;
  dst[((c * 8 + bb) * L + l) * 1024 + n] = f2bf(src[i]);
}

// ---------------------------------------------------------------------------
// Start conv: price [B,N,30,4] -> Xcl[((b*N+n)*32 + l)*128 + c], left-pad 2
// ---------------------------------------------------------------------------
__global__ void k_start(const float* __restrict__ price, const float* __restrict__ w,
                        const float* __restrict__ bvec, float* __restrict__ X, long total) {
  long i = blockIdx.x * (long)blockDim.x + threadIdx.x;
  if (i >= total) return;
  int c = (int)(i & 127);
  long p = i >> 7;
  int l = (int)(p & 31);
  long bn = p >> 5;
  int t = l - 2;
  float acc = bvec[c];
  if (t >= 0 && t < 30) {
    const float* pr = price + (bn * 30 + t) * 4;
    const float* wc = w + c * 4;
    acc += pr[0]*wc[0] + pr[1]*wc[1] + pr[2]*wc[2] + pr[3]*wc[3];
  }
  X[i] = acc;
}

// adpT[n][v] = softmax_v(relu(nv[v]*nv[n]))   (bf16, v-contiguous)
__global__ void k_adj(const float* __restrict__ nv, unsigned short* __restrict__ adpT) {
  int n = blockIdx.x, t = threadIdx.x;
  float xn = nv[n];
  __shared__ float red[256];
  float mx = 0.f;
  for (int i = t; i < 1024; i += 256) mx = fmaxf(mx, fmaxf(nv[i] * xn, 0.f));
  red[t] = mx; __syncthreads();
  for (int o = 128; o > 0; o >>= 1) { if (t < o) red[t] = fmaxf(red[t], red[t + o]); __syncthreads(); }
  mx = red[0]; __syncthreads();
  float s = 0.f;
  for (int i = t; i < 1024; i += 256) s += __expf(fmaxf(nv[i] * xn, 0.f) - mx);
  red[t] = s; __syncthreads();
  for (int o = 128; o > 0; o >>= 1) { if (t < o) red[t] += red[t + o]; __syncthreads(); }
  float inv = 1.f / red[0];
  for (int i = t; i < 1024; i += 256)
    adpT[(long)n * 1024 + i] = f2bf(__expf(fmaxf(nv[i] * xn, 0.f) - mx) * inv);
}

// ---------------------------------------------------------------------------
// Spatial attention small projections (G channel-last; p* layouts [n,b,.])
// ---------------------------------------------------------------------------
__global__ void k_p1a(const float* __restrict__ G, const float* __restrict__ w1,
                      float* __restrict__ p1a, int L, long total) {
  long i = blockIdx.x * (long)blockDim.x + threadIdx.x;
  if (i >= total) return;                         // i = n*1024 + b*128 + c
  int c = (int)(i & 127);
  long r = i >> 7;
  int b = (int)(r & 7);
  long n = r >> 3;
  const float* xp = G + (((long)b * 1024 + n) * L) * 128 + c;
  float s = 0.f;
  for (int l = 0; l < L; ++l) s += xp[(long)l * 128] * w1[l];
  p1a[i] = s;
}

__global__ void k_p1(const float* __restrict__ p1a, const float* __restrict__ w2,
                     float* __restrict__ p1, int L, long total) {
  long i = blockIdx.x * (long)blockDim.x + threadIdx.x;
  if (i >= total) return;                         // i = (n*8+b)*L + l
  int l = (int)(i % L);
  long r = i / L;
  const float* pa = p1a + r * 128;
  const float* wr = w2 + (long)l * 128;
  float s = 0.f;
  for (int c = 0; c < 128; ++c) s += pa[c] * wr[c];
  p1[i] = s;
}

__global__ void k_p2(const float* __restrict__ G, const float* __restrict__ w3,
                     float* __restrict__ p2, int L, long total) {
  long i = blockIdx.x * (long)blockDim.x + threadIdx.x;
  if (i >= total) return;                         // i = (n*8+b)*L + l
  int l = (int)(i % L);
  long r = i / L;
  int b = (int)(r & 7);
  long n = r >> 3;
  const float* xp = G + (((long)b * 1024 + n) * L + l) * 128;
  float s = 0.f;
  for (int c = 0; c < 128; ++c) s += xp[c] * w3[c];
  p2[i] = s;
}

// s[(b*N+n), m] = relu(<p1[n,b,:], p2[m,b,:]>)  (bf16, m-contiguous)
__global__ void k_sbmm(const float* __restrict__ p1, const float* __restrict__ p2,
                       unsigned short* __restrict__ sbf, int L, long total) {
  long i = blockIdx.x * (long)blockDim.x + threadIdx.x;
  if (i >= total) return;                         // i = (b*1024+n)*1024 + m
  int m = (int)(i & 1023);
  long r = i >> 10;
  int n = (int)(r & 1023);
  int b = (int)(r >> 10);
  const float* a = p1 + ((long)n * 8 + b) * L;
  const float* c = p2 + ((long)m * 8 + b) * L;
  float s = 0.f;
  for (int l = 0; l < L; ++l) s += a[l] * c[l];
  sbf[i] = f2bf(fmaxf(s, 0.f));
}

// softmax over rows k of sT [1024, cols]; write transposed bf16 STbf[j*1024+k]
__global__ void k_softmax_colT(const float* __restrict__ sT, unsigned short* __restrict__ STbf,
                               long cols) {
  long j = blockIdx.x;
  int t = threadIdx.x;
  __shared__ float red[256];
  float mx = -1e30f;
  for (int k = t; k < 1024; k += 256) mx = fmaxf(mx, sT[(long)k * cols + j]);
  red[t] = mx; __syncthreads();
  for (int o = 128; o > 0; o >>= 1) { if (t < o) red[t] = fmaxf(red[t], red[t + o]); __syncthreads(); }
  mx = red[0]; __syncthreads();
  float su = 0.f;
  for (int k = t; k < 1024; k += 256) su += __expf(sT[(long)k * cols + j] - mx);
  red[t] = su; __syncthreads();
  for (int o = 128; o > 0; o >>= 1) { if (t < o) red[t] += red[t + o]; __syncthreads(); }
  float inv = 1.f / red[0];
  for (int k = t; k < 1024; k += 256)
    STbf[j * 1024 + k] = f2bf(__expf(sT[(long)k * cols + j] - mx) * inv);
}

__global__ void k_add(float* __restrict__ X, const float* __restrict__ Sx,
                      const float* __restrict__ R, long total) {
  long i = blockIdx.x * (long)blockDim.x + threadIdx.x;
  if (i < total) X[i] = Sx[i] + R[i];
}

// final x[...,0] (channel-last) -> xl[(n*8+b)*128 + h]
__global__ void k_xl(const float* __restrict__ X, float* __restrict__ xl, long total) {
  long i = blockIdx.x * (long)blockDim.x + threadIdx.x;
  if (i >= total) return;                         // i = (n*8+b)*128 + h
  int h = (int)(i & 127);
  long r = i >> 7;
  int b = (int)(r & 7);
  long n = r >> 3;
  xl[i] = X[(((long)b * 1024 + n)) * 128 + h];
}

// head: score[b,n] = out_b + sum_h v[(n*8+b)*128+h]*ow[h]; softmax over n
__global__ void k_head(const float* __restrict__ v, const float* __restrict__ ow,
                       const float* __restrict__ ob, float* __restrict__ out) {
  int b = blockIdx.x, t = threadIdx.x;
  __shared__ float sc[1024];
  __shared__ float red[256];
  for (int n = t; n < 1024; n += 256) {
    const float* vp = v + ((long)n * 8 + b) * 128;
    float s = ob[0];
    for (int h = 0; h < 128; ++h) s += vp[h] * ow[h];
    sc[n] = s;
  }
  __syncthreads();
  float mx = -1e30f;
  for (int n = t; n < 1024; n += 256) mx = fmaxf(mx, sc[n]);
  red[t] = mx; __syncthreads();
  for (int o = 128; o > 0; o >>= 1) { if (t < o) red[t] = fmaxf(red[t], red[t + o]); __syncthreads(); }
  mx = red[0]; __syncthreads();
  float su = 0.f;
  for (int n = t; n < 1024; n += 256) su += __expf(sc[n] - mx);
  red[t] = su; __syncthreads();
  for (int o = 128; o > 0; o >>= 1) { if (t < o) red[t] += red[t + o]; __syncthreads(); }
  float inv = 1.f / red[0];
  for (int n = t; n < 1024; n += 256) out[b * 1024 + n] = __expf(sc[n] - mx) * inv;
}

// ---------------------------------------------------------------------------
// Host orchestration
// ---------------------------------------------------------------------------
enum {
  IN_PRICE = 0, IN_START_W = 1, IN_START_B = 2, IN_BN_START_G = 3, IN_BN_START_B = 4,
  IN_NODEVEC = 5,
  IN_TCN_W = 6, IN_TCN_B = 11, IN_TCN_BN_G = 16, IN_TCN_BN_B = 21,
  IN_RES_W = 26, IN_RES_B = 31, IN_GCN_W = 36, IN_GCN_B = 41,
  IN_SAN_W1 = 46, IN_SAN_W2 = 51, IN_SAN_W3 = 56, IN_SAN_VW = 61, IN_SAN_VB = 66,
  IN_SAN_G1 = 71, IN_SAN_B1 = 76, IN_SAN_G2 = 81, IN_SAN_B2 = 86,
  IN_SAN_G3 = 91, IN_SAN_B3 = 96,
  IN_BN_G = 101, IN_BN_B = 106,
  IN_ASU_BN_G = 111, IN_ASU_BN_B = 112,
  IN_LIN1_W = 113, IN_LIN1_B = 114, IN_OUTLIN_W = 115, IN_OUTLIN_B = 116,
  IN_OUTBN_G = 117, IN_OUTBN_B = 118, IN_OUT_W = 119, IN_OUT_B = 120
};

extern "C" void kernel_launch(void* const* d_in, const int* in_sizes, int n_in,
                              void* d_out, int out_size, void* d_ws, size_t ws_size,
                              hipStream_t stream) {
  (void)in_sizes; (void)n_in; (void)out_size; (void)ws_size;
  const int Bb = 8, Nn = 1024, Hh = 128;
  const int DIL[5] = {1, 2, 4, 8, 16};
  const long ACT = (long)Hh * Bb * Nn * 32;       // max activation element count

  auto F = [&](int i) { return (const float*)d_in[i]; };

  // ---- workspace carve-out -------------------------------------------------
  char* w = (char*)d_ws;
  size_t off = 0;
  auto allocF = [&](long n) { off = (off + 255) & ~(size_t)255;
                              float* p = (float*)(w + off); off += (size_t)n * 4; return p; };
  auto allocU = [&](long n) { off = (off + 255) & ~(size_t)255;
                              unsigned short* p = (unsigned short*)(w + off); off += (size_t)n * 2; return p; };

  float* Xcl  = allocF(ACT);  float* Tcl  = allocF(ACT);
  float* X1nl = allocF(ACT);  float* X2nl = allocF(ACT);
  float* Gcl  = allocF(ACT);  float* Scl  = allocF(ACT);
  float* Rcl  = allocF(ACT);
  unsigned short* Xbfc  = allocU(ACT);
  unsigned short* Tbfc  = allocU(ACT);
  unsigned short* Tbfn  = allocU(ACT);
  unsigned short* X1bfc = allocU(ACT);
  unsigned short* X1bfn = allocU(ACT);
  unsigned short* X2bfc = allocU(ACT);
  unsigned short* Gbfn  = allocU(ACT);
  unsigned short* adpT = allocU((long)Nn * Nn);
  unsigned short* vwbf = allocU((long)Nn * Nn);
  unsigned short* wres = allocU(Hh * Hh);
  unsigned short* wt0  = allocU(Hh * Hh);
  unsigned short* wt1  = allocU(Hh * Hh);
  unsigned short* wgcn = allocU(Hh * 3 * Hh);
  unsigned short* wl1  = allocU(Hh * Hh);
  unsigned short* wl2  = allocU(Hh * Hh);
  float* p1a = allocF((long)Nn * Bb * Hh);
  float* p1  = allocF((long)Nn * Bb * 32);
  float* p2  = allocF((long)Nn * Bb * 32);
  unsigned short* sbf  = allocU((long)Bb * Nn * Nn);
  float* sT  = allocF((long)Nn * Bb * Nn);
  unsigned short* STbf = allocU((long)Bb * Nn * Nn);
  float* xl  = allocF((long)Nn * Bb * Hh);
  unsigned short* xlbf = allocU((long)Nn * Bb * Hh);
  float* y1  = allocF((long)Nn * Bb * Hh);
  unsigned short* y1bf = allocU((long)Nn * Bb * Hh);
  float* vo  = allocF((long)Nn * Bb * Hh);
  float* sums = allocF(256);
  float* mv   = allocF(256);
  float* cmst = allocF(2048);

  // ---- launch helpers ------------------------------------------------------
  auto gemm = [&](float* D, const unsigned short* A, const unsigned short* Bm,
                  int Mr, int Nc, int K, long lda,
                  int bJD, long bJO, long bJI, long bJS,
                  long dRS, int dJD, long dJO, long dJI, long dJS,
                  const float* bias, int accum, int act) {
    int tiles = (Mr >> 4) * (Nc >> 6);
    int blocks = (tiles + 3) >> 2;                // 4 waves / 128-thread block
    k_wmma_gemm<<<blocks, 128, 0, stream>>>(D, A, Bm, Mr, Nc, K, lda,
        bJD, bJO, bJI, bJS, dRS, dJD, dJO, dJI, dJS, bias, accum, act);
  };
  auto cvt = [&](unsigned short* dst, const float* src, long n, int stride) {
    k_f2bf<<<(int)((n + 255) / 256), 256, 0, stream>>>(dst, src, n, stride);
  };
  auto bncl = [&](float* y, const float* x, const float* g, const float* b,
                  long P, int L, int act, unsigned short* bfcl, unsigned short* bfnl) {
    k_zero256<<<1, 256, 0, stream>>>(sums);
    k_bncl_stats<<<(int)((P + BNCHUNK - 1) / BNCHUNK), 256, 0, stream>>>(x, P, sums);
    k_bn_fin<<<1, 128, 0, stream>>>(sums, P, mv);
    long total = P * 128;
    k_bncl_apply<<<(int)((total + 255) / 256), 256, 0, stream>>>(
        y, x, mv, g, b, L, total, act, bfcl, bfnl);
  };
  auto bncm = [&](float* y, const float* x, const float* g, const float* b,
                  int C, long M, unsigned short* bf) {
    k_bncm_stats<<<C, 256, 0, stream>>>(x, M, cmst, cmst + C);
    long total = (long)C * M;
    k_bncm_apply<<<(int)((total + 255) / 256), 256, 0, stream>>>(
        y, x, cmst, cmst + C, g, b, M, total, bf);
  };

  // ---- start conv + bn_start ----------------------------------------------
  {
    long total = (long)Hh * Bb * Nn * 32;
    k_start<<<(int)((total + 255) / 256), 256, 0, stream>>>(
        F(IN_PRICE), F(IN_START_W), F(IN_START_B), Xcl, total);
    bncl(Xcl, Xcl, F(IN_BN_START_G), F(IN_BN_START_B), (long)Bb * Nn * 32, 32, 0, Xbfc, nullptr);
  }
  k_adj<<<Nn, 256, 0, stream>>>(F(IN_NODEVEC), adpT);
  cvt(wl1, F(IN_LIN1_W), Hh * Hh, 1);
  cvt(wl2, F(IN_OUTLIN_W), Hh * Hh, 1);

  // ---- layer loop ----------------------------------------------------------
  int Lin = 32;
  for (int i = 0; i < 5; ++i) {
    const int d = DIL[i];
    const int Lout = Lin - d;
    const long Pout = (long)Bb * Nn * Lout;       // output positions
    const int Mout = (int)Pout;                   // GEMM columns (channel convs)
    const long Eout = Pout * Hh;                  // output elements

    cvt(wres, F(IN_RES_W + i), Hh * Hh, 1);
    cvt(wt0,  F(IN_TCN_W + i),     Hh * Hh, 2);   // tap 0
    cvt(wt1,  F(IN_TCN_W + i) + 1, Hh * Hh, 2);   // tap 1
    cvt(wgcn, F(IN_GCN_W + i), Hh * 3 * Hh, 1);
    cvt(vwbf, F(IN_SAN_VW + i), (long)Nn * Nn, 1);

    // residual = res_w @ x[..., d:]   (shift folded into column map)
    gemm(Rcl, wres, Xbfc, Hh, Mout, Hh, Hh,
         Lout, (long)Lin * 128, 128, (long)d * 128,
         1, 1, 128, 1, 0, F(IN_RES_B + i), 0, 0);

    // dilated TCN: w0 @ x[..., :-d] + w1 @ x[..., d:]  -> relu
    gemm(Tcl, wt0, Xbfc, Hh, Mout, Hh, Hh,
         Lout, (long)Lin * 128, 128, 0,
         1, 1, 128, 1, 0, F(IN_TCN_B + i), 0, 0);
    gemm(Tcl, wt1, Xbfc, Hh, Mout, Hh, Hh,
         Lout, (long)Lin * 128, 128, (long)d * 128,
         1, 1, 128, 1, 0, nullptr, 1, 1);
    bncl(Tcl, Tcl, F(IN_TCN_BN_G + i), F(IN_TCN_BN_B + i), Pout, Lout, 0, Tbfc, Tbfn);

    // diffusion: x1 = adp^T contraction over nodes; x2 = adp(adp(x))
    const int NcN = Hh * Bb * Lout;               // columns (c,b,l) ; %64==0
    gemm(X1nl, adpT, Tbfn, Nn, NcN, Nn, Nn,
         1, 1024, 1, 0,
         1, 1, 1024, 1, 0, nullptr, 0, 0);
    k_cvt_nl<<<(int)((Eout + 255) / 256), 256, 0, stream>>>(X1nl, X1bfn, X1bfc, Lout, Eout);
    gemm(X2nl, adpT, X1bfn, Nn, NcN, Nn, Nn,
         1, 1024, 1, 0,
         1, 1, 1024, 1, 0, nullptr, 0, 0);
    k_cvt_nl<<<(int)((Eout + 255) / 256), 256, 0, stream>>>(X2nl, nullptr, X2bfc, Lout, Eout);

    // gcn: W[:,0:128]@x + W[:,128:256]@x1 + W[:,256:384]@x2 + b
    gemm(Gcl, wgcn,            Tbfc,  Hh, Mout, Hh, 3 * Hh,
         1, 128, 1, 0,  1, 1, 128, 1, 0, F(IN_GCN_B + i), 0, 0);
    gemm(Gcl, wgcn + Hh,       X1bfc, Hh, Mout, Hh, 3 * Hh,
         1, 128, 1, 0,  1, 1, 128, 1, 0, nullptr, 1, 0);
    gemm(Gcl, wgcn + 2 * Hh,   X2bfc, Hh, Mout, Hh, 3 * Hh,
         1, 128, 1, 0,  1, 1, 128, 1, 0, nullptr, 1, 0);
    k_cvt_cl2nl<<<(int)((Eout + 255) / 256), 256, 0, stream>>>(Gcl, Gbfn, Lout, Eout);

    // ---- spatial attention ------------------------------------------------
    long tp = (long)Nn * Bb * Hh;
    k_p1a<<<(int)((tp + 255) / 256), 256, 0, stream>>>(Gcl, F(IN_SAN_W1 + i), p1a, Lout, tp);
    bncm(p1a, p1a, F(IN_SAN_G1 + i), F(IN_SAN_B1 + i), Nn, (long)Bb * Hh, nullptr);
    long tl = (long)Nn * Bb * Lout;
    k_p1<<<(int)((tl + 255) / 256), 256, 0, stream>>>(p1a, F(IN_SAN_W2 + i), p1, Lout, tl);
    bncm(p1, p1, F(IN_SAN_G2 + i), F(IN_SAN_B2 + i), Nn, (long)Bb * Lout, nullptr);
    k_p2<<<(int)((tl + 255) / 256), 256, 0, stream>>>(Gcl, F(IN_SAN_W3 + i), p2, Lout, tl);
    bncm(p2, p2, F(IN_SAN_G3 + i), F(IN_SAN_B3 + i), Nn, (long)Bb * Lout, nullptr);

    long ts = (long)Bb * Nn * Nn;
    k_sbmm<<<(int)((ts + 255) / 256), 256, 0, stream>>>(p1, p2, sbf, Lout, ts);

    // sT[k, (b,n)] = vb[k] + sum_m vw[k,m] * s[(b,n), m]
    gemm(sT, vwbf, sbf, Nn, Bb * Nn, Nn, Nn,
         1, 1024, 1, 0,
         (long)Bb * Nn, 1, 1, 1, 0, F(IN_SAN_VB + i), 0, 0);
    k_softmax_colT<<<Bb * Nn, 256, 0, stream>>>(sT, STbf, (long)Bb * Nn);

    // x = S @ x  (per batch; rows n, contraction over nodes m)
    for (int b = 0; b < Bb; ++b) {
      gemm(Scl, STbf + (long)b * Nn * Nn, Gbfn, Nn, Hh * Lout, Nn, Nn,
           Lout, (long)8 * Lout * 1024, 1024, (long)b * Lout * 1024,
           (long)Lout * 128, Lout, 1, 128, (long)b * Nn * Lout * 128,
           nullptr, 0, 0);
    }

    // x += residual; bn -> next layer input (channel-last bf16 mirror)
    k_add<<<(int)((Eout + 255) / 256), 256, 0, stream>>>(Xcl, Scl, Rcl, Eout);
    bncl(Xcl, Xcl, F(IN_BN_G + i), F(IN_BN_B + i), Pout, Lout, 0, Xbfc, nullptr);

    Lin = Lout;
  }

  // ---- head ----------------------------------------------------------------
  long tx = (long)Nn * Bb * Hh;
  k_xl<<<(int)((tx + 255) / 256), 256, 0, stream>>>(Xcl, xl, tx);
  bncm(xl, xl, F(IN_ASU_BN_G), F(IN_ASU_BN_B), Nn, (long)Bb * Hh, xlbf);

  gemm(y1, wl1, xlbf, Hh, Nn * Bb, Hh, Hh,
       1, 128, 1, 0,
       1, 1, 128, 1, 0, F(IN_LIN1_B), 0, 0);
  cvt(y1bf, y1, tx, 1);

  gemm(vo, wl2, y1bf, Hh, Nn * Bb, Hh, Hh,
       1, 128, 1, 0,
       1, 1, 128, 1, 0, F(IN_OUTLIN_B), 0, 0);
  bncl(vo, vo, F(IN_OUTBN_G), F(IN_OUTBN_B), (long)Nn * Bb, 1, 2 /*tanh*/, nullptr, nullptr);

  k_head<<<Bb, 256, 0, stream>>>(vo, F(IN_OUT_W), F(IN_OUT_B), (float*)d_out);
}
